// BIPA_MultiHeadAttention_45062796869730
// MI455X (gfx1250) — matrix operand order repaired and checked by reading, not compiled, not run on hardware
//
#include <hip/hip_runtime.h>
#include <hip/hip_bf16.h>

#define DIMX   768
#define NHEAD  12
#define HD     64
#define BATCH  32
#define SEQ    577
#define SEQP   608              // padded seq length (mult of 16, covers key tiles to 607)
#define MROWS  (BATCH * SEQ)    // 18464

typedef _Float16 v16h __attribute__((ext_vector_type(16)));
typedef _Float16 v8h  __attribute__((ext_vector_type(8)));
typedef float    v8f  __attribute__((ext_vector_type(8)));

union AFrag { v16h v; v8h h[2]; };

static __device__ __forceinline__ v8f wmma_f32_16x16x32_f16(v16h a, v16h b, v8f c) {
  return __builtin_amdgcn_wmma_f32_16x16x32_f16(false, a, false, b, (short)0, c, false, false);
}

// ---------------------------------------------------------------------------
// f32 -> f16 conversion (grid-stride)
// ---------------------------------------------------------------------------
__global__ void k_cvt_f32_to_f16(const float* __restrict__ src,
                                 _Float16* __restrict__ dst, size_t n) {
  size_t i = (size_t)blockIdx.x * blockDim.x + threadIdx.x;
  size_t stride = (size_t)gridDim.x * blockDim.x;
  for (; i < n; i += stride) dst[i] = (_Float16)src[i];
}

// ---------------------------------------------------------------------------
// 64(M) x 64(N) per-wave WMMA GEMM core:  C += A[64xK] * W[64xK]^T
// 16 WMMAs per K-step against 16 b128 loads (1.0 load/WMMA).
// A-frag: lane holds row (lane%16); 16 contiguous halves at (lane/16)*16
// B-frag: lane holds col (lane%16) = W row; 16 contiguous halves at (lane/16)*16
// ---------------------------------------------------------------------------
static __device__ __forceinline__ void gemm_64x64(
    const _Float16* const Arow[4],        // row (m_base + mt*16 + lane%16), clamped
    const _Float16* const Wrow[4],        // row (n_base + nt*16 + lane%16)
    int K, int hi16, v8f acc[4][4]) {
  for (int k0 = 0; k0 < K; k0 += 32) {
    AFrag a[4];
#pragma unroll
    for (int mt = 0; mt < 4; ++mt) {
      a[mt].v = *(const v16h*)(Arow[mt] + k0 + hi16 * 16);
    }
    v16h b[4];
#pragma unroll
    for (int nt = 0; nt < 4; ++nt)
      b[nt] = *(const v16h*)(Wrow[nt] + k0 + hi16 * 16);
    if (k0 + 32 < K) {                       // gfx1250 global_prefetch_b8 (near)
      __builtin_prefetch(Arow[0] + k0 + 96, 0, 3);
      __builtin_prefetch(Arow[2] + k0 + 96, 0, 3);
      __builtin_prefetch(Wrow[0] + k0 + 96, 0, 3);
      __builtin_prefetch(Wrow[2] + k0 + 96, 0, 3);
    }
#pragma unroll
    for (int mt = 0; mt < 4; ++mt)
#pragma unroll
      for (int nt = 0; nt < 4; ++nt)
        acc[mt][nt] = wmma_f32_16x16x32_f16(a[mt].v, b[nt], acc[mt][nt]);
  }
}

// ---------------------------------------------------------------------------
// Stage 1: qkv = x @ qkv_w^T + qkv_b ; scatter into Q,K ([B,H,Lp,64]) and
// V transposed ([B,H,64,Lp]); Q pre-scaled by hd^-0.5 = 0.125.
// grid: (ceil(MROWS/64)=289, 2304/256=9), block 128 (4 waves, each 64x64)
// ---------------------------------------------------------------------------
__global__ __launch_bounds__(128) void k_qkv_gemm(
    const _Float16* __restrict__ xh, const _Float16* __restrict__ wh,
    const float* __restrict__ bias,
    _Float16* __restrict__ qh, _Float16* __restrict__ kh,
    _Float16* __restrict__ vth) {
  const int lane = threadIdx.x & 31, wid = threadIdx.x >> 5;
  const int hi16 = lane >> 4, l16 = lane & 15;
  const int m_base = blockIdx.x * 64;
  const int n_base = blockIdx.y * 256 + wid * 64;

  const _Float16* Arow[4];
#pragma unroll
  for (int mt = 0; mt < 4; ++mt) {
    int r = m_base + mt * 16 + l16;
    if (r > MROWS - 1) r = MROWS - 1;          // clamp (loads only; stores guarded)
    Arow[mt] = xh + (size_t)r * DIMX;
  }
  const _Float16* Wrow[4];
#pragma unroll
  for (int nt = 0; nt < 4; ++nt)
    Wrow[nt] = wh + (size_t)(n_base + nt * 16 + l16) * DIMX;

  v8f acc[4][4] = {};
  gemm_64x64(Arow, Wrow, DIMX, hi16, acc);

#pragma unroll
  for (int nt = 0; nt < 4; ++nt) {
    const int n = n_base + nt * 16 + l16;
    const float bv = bias[n];
    const int part = n / DIMX;            // 0=q 1=k 2=v
    const int rem = n - part * DIMX;
    const int head = rem >> 6;
    const int d = rem & 63;
#pragma unroll
    for (int mt = 0; mt < 4; ++mt) {
#pragma unroll
      for (int i = 0; i < 8; ++i) {
        const int m = m_base + mt * 16 + i + hi16 * 8;
        if (m < MROWS) {
          const int bidx = m / SEQ;
          const int l = m - bidx * SEQ;
          const int bh = bidx * NHEAD + head;
          const float v = acc[mt][nt][i] + bv;
          if (part == 0)
            qh[((size_t)bh * SEQP + l) * HD + d] = (_Float16)(v * 0.125f);
          else if (part == 1)
            kh[((size_t)bh * SEQP + l) * HD + d] = (_Float16)v;
          else
            vth[((size_t)bh * HD + d) * SEQP + l] = (_Float16)v;
        }
      }
    }
  }
}

// ---------------------------------------------------------------------------
// Stage 2: flash attention. One wave = 16 query rows x full hd=64.
// grid: (ceil(37/4)=10, B*H=384), block 128.
// ---------------------------------------------------------------------------
__global__ __launch_bounds__(128) void k_attn(
    const _Float16* __restrict__ qh, const _Float16* __restrict__ kh,
    const _Float16* __restrict__ vth, const float* __restrict__ mb,
    const float* __restrict__ alpha_p, _Float16* __restrict__ aoh) {
  __shared__ _Float16 lds_p[4][16 * 32];   // per-wave P staging (C->A transpose)
  const int lane = threadIdx.x & 31, wid = threadIdx.x >> 5;
  const int hi16 = lane >> 4, l16 = lane & 15;
  const int bh = blockIdx.y;
  const int bat = bh / NHEAD;
  const int head = bh - bat * NHEAD;
  const int l0 = (blockIdx.x * 4 + wid) * 16;
  if (l0 >= SEQ) return;                   // no barriers used -> safe early exit
  const float alpha = alpha_p[0];

  const _Float16* Q = qh + (size_t)bh * SEQP * HD;
  const _Float16* Kp = kh + (size_t)bh * SEQP * HD;
  const _Float16* VT = vth + (size_t)bh * HD * SEQP;

  // Q A-fragments over the two K=32 chunks of hd
  AFrag aq[2];
  {
    const _Float16* qrow = Q + (size_t)(l0 + l16) * HD;
#pragma unroll
    for (int kk = 0; kk < 2; ++kk) {
      aq[kk].v = *(const v16h*)(qrow + kk * 32 + hi16 * 16);
    }
  }

  v8f o[4] = {};
  float mrow[8], srow[8];
#pragma unroll
  for (int i = 0; i < 8; ++i) { mrow[i] = -__builtin_inff(); srow[i] = 0.f; }

  for (int j0 = 0; j0 < SEQ; j0 += 32) {
    // ---- S = Q @ K^T (16 x 32 tile via two 16x16 frags) ----
    v8f s[2];
#pragma unroll
    for (int jt = 0; jt < 2; ++jt) {
      v8f sv = {0.f, 0.f, 0.f, 0.f, 0.f, 0.f, 0.f, 0.f};
#pragma unroll
      for (int kk = 0; kk < 2; ++kk) {
        const _Float16* kr =
            Kp + (size_t)(j0 + jt * 16 + l16) * HD + kk * 32 + hi16 * 16;
        v16h bk = *(const v16h*)kr;
        sv = wmma_f32_16x16x32_f16(aq[kk].v, bk, sv);
      }
      s[jt] = sv;
    }
    // ---- additive bias alpha*mb (col j) + length mask ----
#pragma unroll
    for (int jt = 0; jt < 2; ++jt) {
      const int j = j0 + jt * 16 + l16;
      if (j < SEQ) {
        const float bias = (j == 0) ? 0.f : mb[(size_t)bat * (SEQ - 1) + (j - 1)];
        const float ab = alpha * bias;
#pragma unroll
        for (int i = 0; i < 8; ++i) s[jt][i] += ab;
      } else {
#pragma unroll
        for (int i = 0; i < 8; ++i) s[jt][i] = -__builtin_inff();
      }
    }
    // ---- online softmax (row reductions across the 16-lane half) ----
    float pl0[8], pl1[8];
#pragma unroll
    for (int i = 0; i < 8; ++i) {
      float cm = fmaxf(s[0][i], s[1][i]);
      cm = fmaxf(cm, __shfl_xor(cm, 1, 32));
      cm = fmaxf(cm, __shfl_xor(cm, 2, 32));
      cm = fmaxf(cm, __shfl_xor(cm, 4, 32));
      cm = fmaxf(cm, __shfl_xor(cm, 8, 32));
      const float nm = fmaxf(mrow[i], cm);
      const float sc = __expf(mrow[i] - nm);
      const float p0 = __expf(s[0][i] - nm);
      const float p1 = __expf(s[1][i] - nm);
      float ps = p0 + p1;
      ps += __shfl_xor(ps, 1, 32);
      ps += __shfl_xor(ps, 2, 32);
      ps += __shfl_xor(ps, 4, 32);
      ps += __shfl_xor(ps, 8, 32);
      srow[i] = srow[i] * sc + ps;
      mrow[i] = nm;
      pl0[i] = p0; pl1[i] = p1;
#pragma unroll
      for (int dt = 0; dt < 4; ++dt) o[dt][i] *= sc;
    }
    // ---- P: C-layout -> A-layout via per-wave LDS ----
    _Float16* pp = lds_p[wid];
#pragma unroll
    for (int i = 0; i < 8; ++i) {
      const int r = i + hi16 * 8;
      pp[r * 32 + l16] = (_Float16)pl0[i];
      pp[r * 32 + 16 + l16] = (_Float16)pl1[i];
    }
    asm volatile("s_wait_dscnt 0" ::: "memory");
    AFrag ap;
    ap.v = *(const v16h*)(pp + l16 * 32 + hi16 * 16);
    // ---- O += P @ V (V transposed: contiguous along keys) ----
#pragma unroll
    for (int dt = 0; dt < 4; ++dt) {
      const _Float16* vr = VT + (size_t)(dt * 16 + l16) * SEQP + j0 + hi16 * 16;
      v16h bv = *(const v16h*)vr;
      o[dt] = wmma_f32_16x16x32_f16(ap.v, bv, o[dt]);
    }
  }

  // ---- normalize, store as f16 [B, L, H*hd] for the proj GEMM ----
#pragma unroll
  for (int i = 0; i < 8; ++i) {
    const float r = 1.0f / srow[i];
    const int l = l0 + i + hi16 * 8;
    if (l < SEQ) {
      const size_t row = ((size_t)bat * SEQ + l) * DIMX + head * HD;
#pragma unroll
      for (int dt = 0; dt < 4; ++dt)
        aoh[row + dt * 16 + l16] = (_Float16)(o[dt][i] * r);
    }
  }
}

// ---------------------------------------------------------------------------
// Stage 3: out = attn_out @ proj_w^T + proj_b (f32 output)
// grid: (ceil(MROWS/64)=289, 768/256=3), block 128
// ---------------------------------------------------------------------------
__global__ __launch_bounds__(128) void k_proj_gemm(
    const _Float16* __restrict__ aoh, const _Float16* __restrict__ wh,
    const float* __restrict__ bias, float* __restrict__ out) {
  const int lane = threadIdx.x & 31, wid = threadIdx.x >> 5;
  const int hi16 = lane >> 4, l16 = lane & 15;
  const int m_base = blockIdx.x * 64;
  const int n_base = blockIdx.y * 256 + wid * 64;

  const _Float16* Arow[4];
#pragma unroll
  for (int mt = 0; mt < 4; ++mt) {
    int r = m_base + mt * 16 + l16;
    if (r > MROWS - 1) r = MROWS - 1;
    Arow[mt] = aoh + (size_t)r * DIMX;
  }
  const _Float16* Wrow[4];
#pragma unroll
  for (int nt = 0; nt < 4; ++nt)
    Wrow[nt] = wh + (size_t)(n_base + nt * 16 + l16) * DIMX;

  v8f acc[4][4] = {};
  gemm_64x64(Arow, Wrow, DIMX, hi16, acc);

#pragma unroll
  for (int nt = 0; nt < 4; ++nt) {
    const int n = n_base + nt * 16 + l16;
    const float bv = bias[n];
#pragma unroll
    for (int mt = 0; mt < 4; ++mt) {
#pragma unroll
      for (int i = 0; i < 8; ++i) {
        const int m = m_base + mt * 16 + i + hi16 * 8;
        if (m < MROWS) out[(size_t)m * DIMX + n] = acc[mt][nt][i] + bv;
      }
    }
  }
}

// ---------------------------------------------------------------------------
extern "C" void kernel_launch(void* const* d_in, const int* in_sizes, int n_in,
                              void* d_out, int out_size, void* d_ws,
                              size_t ws_size, hipStream_t stream) {
  const float* x      = (const float*)d_in[0];
  const float* mb     = (const float*)d_in[1];
  const float* qkv_w  = (const float*)d_in[2];
  const float* qkv_b  = (const float*)d_in[3];
  const float* proj_w = (const float*)d_in[4];
  const float* proj_b = (const float*)d_in[5];
  const float* alpha  = (const float*)d_in[6];
  float* out = (float*)d_out;
  (void)in_sizes; (void)n_in; (void)out_size; (void)ws_size;

  char* w = (char*)d_ws;
  size_t off = 0;
  auto take = [&](size_t bytes) -> void* {
    void* p = w + off;
    off += (bytes + 255) & ~(size_t)255;
    return p;
  };
  const size_t qkv_buf_bytes = (size_t)BATCH * NHEAD * SEQP * HD * sizeof(_Float16);
  _Float16* xh    = (_Float16*)take((size_t)MROWS * DIMX * sizeof(_Float16));
  _Float16* wqkvh = (_Float16*)take((size_t)3 * DIMX * DIMX * sizeof(_Float16));
  _Float16* wprjh = (_Float16*)take((size_t)DIMX * DIMX * sizeof(_Float16));
  _Float16* qh    = (_Float16*)take(qkv_buf_bytes);
  _Float16* kh    = (_Float16*)take(qkv_buf_bytes);
  _Float16* vth   = (_Float16*)take(qkv_buf_bytes);
  _Float16* aoh   = (_Float16*)take((size_t)MROWS * DIMX * sizeof(_Float16));

  // zero q/k/vt (contiguous) so padded rows/cols are finite (P=0 * NaN guard)
  hipMemsetAsync(qh, 0, 3 * qkv_buf_bytes, stream);

  k_cvt_f32_to_f16<<<4096, 256, 0, stream>>>(x, xh, (size_t)MROWS * DIMX);
  k_cvt_f32_to_f16<<<2048, 256, 0, stream>>>(qkv_w, wqkvh, (size_t)3 * DIMX * DIMX);
  k_cvt_f32_to_f16<<<1024, 256, 0, stream>>>(proj_w, wprjh, (size_t)DIMX * DIMX);

  k_qkv_gemm<<<dim3((MROWS + 63) / 64, (3 * DIMX) / 256), 128, 0, stream>>>(
      xh, wqkvh, qkv_b, qh, kh, vth);

  const int qtiles = (SEQ + 15) / 16;                 // 37
  k_attn<<<dim3((qtiles + 3) / 4, BATCH * NHEAD), 128, 0, stream>>>(
      qh, kh, vth, mb, alpha, aoh);

  k_proj_gemm<<<dim3((MROWS + 63) / 64, DIMX / 256), 128, 0, stream>>>(
      aoh, wprjh, proj_b, out);
}
